// SelectiveEmbedding_60876866453789
// MI455X (gfx1250) — matrix operand order repaired and checked by reading, not compiled, not run on hardware
//
#include <hip/hip_runtime.h>
#include <hip/hip_bf16.h>

// ---------------------------------------------------------------------------
// MoE top-2 (unweighted combine) for MI455X / gfx1250, wave32 + WMMA bf16.
//   T = 8192 tokens, D = 1024, E = 8 experts, k = 2.
// Grouped-GEMM formulation: 34.4 GFLOP effective, ~100 MB HBM traffic.
// Weights (32 MB) + activations (32 MB) fit in the 192 MB L2 -> tile re-reads
// are L2 hits. bf16 WMMA 16x16x32 is the widest applicable matrix op.
// ---------------------------------------------------------------------------

typedef __attribute__((ext_vector_type(16))) __bf16 v16bf;
typedef __attribute__((ext_vector_type(2)))  __bf16 v2bf;
typedef __attribute__((ext_vector_type(8)))  float  v8f;

#define T_TOK 8192
#define DIM   1024
#define NEXP  8
#define BM    128
#define BN    64
#define BK    32
#define LDST  40   // 40 halves = 80 B row stride: 16B-aligned rows (b128-able),
                   // 20-bank step -> 16 rows hit 16 distinct banks

// ---------------------------------------------------------------------------
// Phase 1: zero output (we accumulate atomically) + per-expert counters.
// ---------------------------------------------------------------------------
__global__ void moe_zero(float* __restrict__ out, int* __restrict__ counts, long n) {
  long i = (long)blockIdx.x * blockDim.x + threadIdx.x;
  long stride = (long)gridDim.x * blockDim.x;
  for (; i < n; i += stride) out[i] = 0.0f;
  if (blockIdx.x == 0 && threadIdx.x < NEXP) counts[threadIdx.x] = 0;
}

// ---------------------------------------------------------------------------
// Phase 2: routing. One wave32 per token; softmax is monotonic and the
// combine is unweighted, so routing = arg-top-2 of the logits.
// ---------------------------------------------------------------------------
__global__ void __launch_bounds__(256)
moe_route(const float* __restrict__ x, const float* __restrict__ gw,
          const float* __restrict__ gb, int* __restrict__ counts,
          int* __restrict__ lists) {
  const int wv   = threadIdx.x >> 5;   // 8 waves per block
  const int lane = threadIdx.x & 31;
  const int t    = blockIdx.x * 8 + wv;

  float acc[NEXP];
#pragma unroll
  for (int e = 0; e < NEXP; ++e) acc[e] = 0.0f;

  const float* xr = x + (size_t)t * DIM;
  for (int d = lane; d < DIM; d += 32) {
    float xv = xr[d];
    const float4* g = (const float4*)(gw + (size_t)d * NEXP);
    float4 g0 = g[0];
    float4 g1 = g[1];
    acc[0] += xv * g0.x; acc[1] += xv * g0.y;
    acc[2] += xv * g0.z; acc[3] += xv * g0.w;
    acc[4] += xv * g1.x; acc[5] += xv * g1.y;
    acc[6] += xv * g1.z; acc[7] += xv * g1.w;
  }

#pragma unroll
  for (int off = 16; off >= 1; off >>= 1) {
#pragma unroll
    for (int e = 0; e < NEXP; ++e)
      acc[e] += __shfl_xor(acc[e], off, 32);
  }

  if (lane == 0) {
    float best1 = -3.4e38f; int e1 = 0;
#pragma unroll
    for (int e = 0; e < NEXP; ++e) {
      float v = acc[e] + gb[e];
      acc[e] = v;
      if (v > best1) { best1 = v; e1 = e; }
    }
    float best2 = -3.4e38f; int e2 = (e1 == 0) ? 1 : 0;
#pragma unroll
    for (int e = 0; e < NEXP; ++e) {
      if (e == e1) continue;
      if (acc[e] > best2) { best2 = acc[e]; e2 = e; }
    }
    int p1 = atomicAdd(&counts[e1], 1);
    lists[e1 * T_TOK + p1] = t;
    int p2 = atomicAdd(&counts[e2], 1);
    lists[e2 * T_TOK + p2] = t;
  }
}

// ---------------------------------------------------------------------------
// Phase 3: grouped GEMM per expert. 128 threads = 4 waves; tile 128x64,
// K stepped by 32. A rows gathered via per-expert token lists (f32 -> bf16
// into LDS, 16B-aligned rows -> b128 stores/loads). B staged TRANSPOSED in
// LDS (paired-K packed dword stores) so each lane's B fragment is one
// contiguous 32 B run (k = 16h..16h+15, same order as A) -> ds_load_b128 x2.
// All 6 fragments (2 A + 4 B) are materialized in distinct registers BEFORE
// the WMMA block so the 12 ds_load_b128 issue as one burst and the 8 WMMAs
// drain behind descending partial s_wait_dscnt.
// ---------------------------------------------------------------------------
__global__ void __launch_bounds__(128)
moe_expert_gemm(const float* __restrict__ x, const float* __restrict__ w,
                const int* __restrict__ counts, const int* __restrict__ lists,
                float* __restrict__ out) {
  const int e   = blockIdx.z;
  const int cnt = counts[e];
  const int m0  = blockIdx.y * BM;
  if (m0 >= cnt) return;                  // uniform early-exit for empty tiles
  const int n0  = blockIdx.x * BN;

  __shared__ int    toks[BM];
  __shared__ __bf16 As[BM][LDST];         // row-major  [m][k]
  __shared__ __bf16 Bt[BN][LDST];         // transposed [n][k]

  const int tid = threadIdx.x;
  {
    int gm = m0 + tid;
    toks[tid] = (gm < cnt) ? lists[e * T_TOK + gm] : -1;
  }
  __syncthreads();

  const int wv   = tid >> 5;              // wave 0..3 -> 32-row M strip
  const int lane = tid & 31;
  const int g    = lane >> 4;             // half-wave group (ISA layouts)
  const int ml   = lane & 15;

  v8f acc[2][4] = {};                     // 2 M-subtiles x 4 N-subtiles

  // B staging: each thread owns a K-row pair x 8 columns -> packed v2bf stores.
  const int bkp  = (tid >> 3) * 2;        // K row pair base: 0,2,..,30
  const int bcol = (tid & 7) * 8;         // 8-column segment
  const float* wbase = w + (size_t)e * DIM * DIM;
  const int mytok = toks[tid];            // A stage: this thread's full row

  // Invalid gather rows are zero for every K-step: zero them ONCE.
  if (mytok < 0) {
#pragma unroll
    for (int q = 0; q < BK; ++q) As[tid][q] = (__bf16)0.0f;
  }

  for (int kk = 0; kk < DIM; kk += BK) {
    // ---- stage A: one gathered token row per thread, f32 -> bf16 ----
    if (mytok >= 0) {
      const float4* src = (const float4*)(x + (size_t)mytok * DIM + kk);
#pragma unroll
      for (int q = 0; q < 8; ++q) {
        float4 v = src[q];
        As[tid][4 * q + 0] = (__bf16)v.x;
        As[tid][4 * q + 1] = (__bf16)v.y;
        As[tid][4 * q + 2] = (__bf16)v.z;
        As[tid][4 * q + 3] = (__bf16)v.w;
      }
    }
    // ---- stage B: coalesced reads of a K-row pair, packed transposed stores ----
    {
      const float* r0 = wbase + (size_t)(kk + bkp + 0) * DIM + n0 + bcol;
      const float* r1 = wbase + (size_t)(kk + bkp + 1) * DIM + n0 + bcol;
      float4 a0 = ((const float4*)r0)[0];
      float4 a1 = ((const float4*)r0)[1];
      float4 b0 = ((const float4*)r1)[0];
      float4 b1 = ((const float4*)r1)[1];
      float c0[8] = {a0.x, a0.y, a0.z, a0.w, a1.x, a1.y, a1.z, a1.w};
      float c1[8] = {b0.x, b0.y, b0.z, b0.w, b1.x, b1.y, b1.z, b1.w};
#pragma unroll
      for (int j = 0; j < 8; ++j) {
        v2bf p;
        p[0] = (__bf16)c0[j];             // K = bkp
        p[1] = (__bf16)c1[j];             // K = bkp+1
        *(v2bf*)&Bt[bcol + j][bkp] = p;   // 4B-aligned packed store
      }
    }
    // ---- prefetch next K-step's global data into cache ----
    if (kk + BK < DIM) {
      if (mytok >= 0) __builtin_prefetch(x + (size_t)mytok * DIM + kk + BK, 0, 3);
      __builtin_prefetch(wbase + (size_t)(kk + BK + bkp) * DIM + n0 + bcol, 0, 3);
    }
    __syncthreads();

    // ---- A fragments (16x32 bf16, ISA 7.12.2): two 16B-aligned b128 chunks ----
    v16bf af[2];
#pragma unroll
    for (int s = 0; s < 2; ++s) {
      const int arow = 32 * wv + 16 * s + ml;
#pragma unroll
      for (int v = 0; v < 8; ++v) {
        int k0 = 16 * g + 2 * v;
        af[s][2 * v + 0] = As[arow][k0 + 0];
        af[s][2 * v + 1] = As[arow][k0 + 1];
      }
    }

    // ---- B fragments: ALL FOUR materialized up-front (distinct registers) ----
    v16bf bfr[4];
#pragma unroll
    for (int c = 0; c < 4; ++c) {
      const int brow = 16 * c + ml;
#pragma unroll
      for (int v = 0; v < 8; ++v) {
        bfr[c][2 * v + 0] = Bt[brow][16 * g + 2 * v + 0];
        bfr[c][2 * v + 1] = Bt[brow][16 * g + 2 * v + 1];
      }
    }

    // ---- 8 WMMAs drain behind partial dscnt waits ----
#pragma unroll
    for (int c = 0; c < 4; ++c) {
#pragma unroll
      for (int s = 0; s < 2; ++s) {
        acc[s][c] = __builtin_amdgcn_wmma_f32_16x16x32_bf16(
            /*neg_a=*/false, af[s], /*neg_b=*/false, bfr[c],
            /*c_mod=*/(short)0, acc[s][c], /*reuse_a=*/false, /*reuse_b=*/false);
      }
    }
    __syncthreads();
  }

  // ---- combine: scatter-accumulate (each token hit by exactly 2 experts) ----
#pragma unroll
  for (int s = 0; s < 2; ++s) {
#pragma unroll
    for (int c = 0; c < 4; ++c) {
#pragma unroll
      for (int r = 0; r < 8; ++r) {
        int row = 32 * wv + 16 * s + r + 8 * g;  // C/D: VGPR r -> M=r (+8 hi half)
        int tok = toks[row];
        if (tok >= 0)
          atomicAdd(out + (size_t)tok * DIM + n0 + 16 * c + ml, acc[s][c][r]);
      }
    }
  }
}

// ---------------------------------------------------------------------------
extern "C" void kernel_launch(void* const* d_in, const int* in_sizes, int n_in,
                              void* d_out, int out_size, void* d_ws, size_t ws_size,
                              hipStream_t stream) {
  const float* x  = (const float*)d_in[0];   // E_symb   [4,2048,1024] f32
  const float* gw = (const float*)d_in[1];   // gate_w   [1024,8]      f32
  const float* gb = (const float*)d_in[2];   // gate_b   [8]           f32
  const float* w  = (const float*)d_in[3];   // expert_w [8,1024,1024] f32
  // d_in[4] is k (== 2), hardcoded.
  float* out = (float*)d_out;

  int* counts = (int*)d_ws;                        // 8 ints
  int* lists  = (int*)((char*)d_ws + 64);          // 8 * 8192 ints (~256 KB)

  const long n_out = (long)T_TOK * DIM;
  moe_zero<<<4096, 256, 0, stream>>>(out, counts, n_out);

  moe_route<<<T_TOK / 8, 256, 0, stream>>>(x, gw, gb, counts, lists);

  dim3 grid(DIM / BN, T_TOK / BM, NEXP);           // worst case: all tokens one expert
  moe_expert_gemm<<<grid, 128, 0, stream>>>(x, w, counts, lists, out);
}
